// RWKV_Tmix_x070_Mose_cxa073_39024072851816
// MI455X (gfx1250) — matrix operand order repaired and checked by reading, not compiled, not run on hardware
//
#include <hip/hip_runtime.h>
#include <cstddef>
#include <cstdint>

// ---- problem constants (match reference) ----
#define B_SZ   4
#define T_SZ   1024
#define C_SZ   1024
#define H_N    16
#define HS_N   64
#define HKV_N  4
#define NREP_N 4
#define M_SZ   (B_SZ * T_SZ)      // 4096 token rows
#define KV_C   (HKV_N * HS_N)     // 256
#define DD_N   64                 // D_DECAY
#define DMV_N  32                 // D_MV
#define EPS_F  0.00064f

typedef __attribute__((ext_vector_type(16))) __bf16 v16bf;
typedef __attribute__((ext_vector_type(8)))  __bf16 v8bf;
typedef __attribute__((ext_vector_type(8)))  float  v8f;

// ------------------------------------------------------------------
// fp32 -> bf16 convert, optional tanh (mode=1)
// ------------------------------------------------------------------
__global__ void k_cvt_bf16(const float* __restrict__ in, __bf16* __restrict__ out,
                           int n, int do_tanh) {
  int i = blockIdx.x * blockDim.x + threadIdx.x;
  if (i >= n) return;
  float v = in[i];
  if (do_tanh) v = tanhf(v);
  out[i] = (__bf16)v;
}

// transpose-convert: in is (R, Ccols) fp32 row-major, out is (Ccols, R) bf16 row-major
__global__ void k_trans_bf16(const float* __restrict__ in, __bf16* __restrict__ out,
                             int R, int Ccols) {
  int i = blockIdx.x * blockDim.x + threadIdx.x;
  if (i >= R * Ccols) return;
  int r = i / Ccols, c = i % Ccols;
  out[(size_t)c * R + r] = (__bf16)in[i];
}

// ------------------------------------------------------------------
// bf16 WMMA GEMM: C[M,N] = A[M,K] @ W[N,K]^T   (A,W bf16 row-major, C fp32)
// Block = 128 threads = 4 waves. Each wave: (16*MSUB) x (16*NSUB) tile, K step 32.
// Branch-free inner loop -> straight-line v_wmma_f32_16x16x32_bf16 sequence.
// Requires: M % (64*MSUB) == 0, N % (16*NSUB) == 0, K % 32 == 0.
// ------------------------------------------------------------------
template <int MSUB, int NSUB>
__global__ __launch_bounds__(128)
void k_gemm_bf16(const __bf16* __restrict__ A, const __bf16* __restrict__ W,
                 float* __restrict__ Cout, int M, int N, int K) {
  const int lane = threadIdx.x & 31;
  const int wave = threadIdx.x >> 5;
  const int half = lane >> 4;      // 0: low half, 1: high half of wave
  const int l16  = lane & 15;
  const int tileM = (blockIdx.x * 4 + wave) * (16 * MSUB);
  const int tileN = blockIdx.y * (16 * NSUB);

  v8f zero = {};
  v8f acc[MSUB][NSUB];
#pragma unroll
  for (int m = 0; m < MSUB; ++m)
#pragma unroll
    for (int n = 0; n < NSUB; ++n) acc[m][n] = zero;

  const __bf16* Abase = A + (size_t)(tileM + l16) * K;
  const __bf16* Wbase = W + (size_t)(tileN + l16) * K;

  for (int k0 = 0; k0 < K; k0 += 32) {
    // A fragment: 16x32 bf16. lane<16 holds K {0..15}, lane>=16 {16..31}
    v16bf af[MSUB];
#pragma unroll
    for (int m = 0; m < MSUB; ++m) {
      union { v16bf v; v8bf h[2]; } u;
      const __bf16* ap = Abase + (size_t)m * 16 * K + k0 + half * 16;
      u.h[0] = *(const v8bf*)(ap);
      u.h[1] = *(const v8bf*)(ap + 8);
      af[m] = u.v;
    }
#pragma unroll
    for (int n = 0; n < NSUB; ++n) {
      // B fragment: 32x16 (KxN). lane = column, element e -> K = k0 + half*16 + e
      v16bf bf = *(const v16bf*)(Wbase + (size_t)n * 16 * K + k0 + half * 16);
#pragma unroll
      for (int m = 0; m < MSUB; ++m) {
        acc[m][n] = __builtin_amdgcn_wmma_f32_16x16x32_bf16(
            false, af[m], false, bf, (short)0, acc[m][n], false, false);
      }
    }
  }

  // C/D layout: VGPR j -> row = j + 8*half, col = l16
#pragma unroll
  for (int m = 0; m < MSUB; ++m)
#pragma unroll
    for (int n = 0; n < NSUB; ++n)
#pragma unroll
      for (int j = 0; j < 8; ++j) {
        int row = tileM + m * 16 + j + 8 * half;
        int col = tileN + n * 16 + l16;
        Cout[(size_t)row * N + col] = acc[m][n][j];
      }
}

// ------------------------------------------------------------------
// 64-thread (one head) LDS tree reduction
// ------------------------------------------------------------------
__device__ __forceinline__ float head_reduce64(float v, volatile float* red, int hs) {
  red[hs] = v; __syncthreads();
  if (hs < 32) red[hs] += red[hs + 32]; __syncthreads();
  if (hs < 16) red[hs] += red[hs + 16]; __syncthreads();
  if (hs <  8) red[hs] += red[hs +  8]; __syncthreads();
  if (hs <  4) red[hs] += red[hs +  4]; __syncthreads();
  if (hs <  2) red[hs] += red[hs +  2]; __syncthreads();
  if (hs <  1) red[hs] += red[hs +  1]; __syncthreads();
  float s = red[0]; __syncthreads();
  return s;
}

// ------------------------------------------------------------------
// Gate / mix kernel. One 64-thread block per (b,t,h).
// In-place: wpre->decay, apre->aa(=-kk), vpre->v(mixed). Also writes kfin, bb(=kk*a).
// ------------------------------------------------------------------
__global__ void k_mix(const float* __restrict__ ksm, const float* __restrict__ vsm,
                      const float* __restrict__ vfirst,
                      float* __restrict__ wpre_dec, float* __restrict__ apre_aa,
                      float* __restrict__ vpre_v,
                      float* __restrict__ kfin, float* __restrict__ bbuf,
                      const float* __restrict__ w0, const float* __restrict__ a0,
                      const float* __restrict__ v0, const float* __restrict__ k_k,
                      const float* __restrict__ k_a) {
  __shared__ float red[64];
  const int bt = blockIdx.x / H_N;          // b*T + t
  const int h  = blockIdx.x % H_N;
  const int hs = threadIdx.x;
  const int c  = h * HS_N + hs;
  const size_t idx  = (size_t)bt * C_SZ + c;
  const int kvh = h / NREP_N;
  const size_t sidx = (size_t)bt * KV_C + kvh * HS_N + hs;

  // decay = exp(-exp(-softplus(-(w0 + dw)) - 0.6))
  float z = w0[c] + wpre_dec[idx];
  float w = -log1pf(expf(-z)) - 0.6f;
  float decay = expf(-expf(w));

  float a_sig = 1.0f / (1.0f + expf(-(a0[c] + apre_aa[idx])));
  float gate  = 1.0f / (1.0f + expf(-(v0[c] + vpre_v[idx])));

  float kr = ksm[sidx];                      // GQA-repeated k
  float vr = vsm[sidx];
  float vm = vr + (vfirst[idx] - vr) * gate; // residual mix with first-layer v

  float kkraw = kr * k_k[c];
  float ss = head_reduce64(kkraw * kkraw, red, hs);
  float nrm = sqrtf(ss);
  nrm = fmaxf(nrm, 1e-12f);
  float kk = kkraw / nrm;

  float kf = kr * (1.0f + (a_sig - 1.0f) * k_a[c]);

  wpre_dec[idx] = decay;
  apre_aa[idx]  = -kk;           // a-input of wkv7
  vpre_v[idx]   = vm;
  kfin[idx]     = kf;
  bbuf[idx]     = kk * a_sig;    // b-input of wkv7
}

// ------------------------------------------------------------------
// RWKV-7 delta-rule recurrence. One 64-thread block per (b,h).
// Thread i owns row i (value dim) of S[64][64] in registers.
// ------------------------------------------------------------------
__global__ void k_wkv7(const float* __restrict__ r, const float* __restrict__ dec,
                       const float* __restrict__ k, const float* __restrict__ v,
                       const float* __restrict__ aa, const float* __restrict__ bb,
                       const float* __restrict__ mask, float* __restrict__ y) {
  const int b = blockIdx.x / H_N;
  const int h = blockIdx.x % H_N;
  const int i = threadIdx.x;

  float S[64];
#pragma unroll
  for (int j = 0; j < 64; ++j) S[j] = 0.0f;

  __shared__ float sr[64], sd[64], sk[64], sv[64], sa[64], sb[64];
  const size_t base = (size_t)b * T_SZ * C_SZ + (size_t)h * HS_N;

  for (int t = 0; t < T_SZ; ++t) {
    const size_t o = base + (size_t)t * C_SZ;
    float m = mask[(size_t)b * T_SZ + t];
    sr[i] = r[o + i];
    sd[i] = (m > 0.0f) ? dec[o + i] : 1.0f;
    sk[i] = k[o + i] * m;
    sv[i] = v[o + i] * m;
    sa[i] = aa[o + i] * m;
    sb[i] = bb[o + i] * m;
    __syncthreads();

    float sacc = 0.0f;
#pragma unroll
    for (int j = 0; j < 64; ++j) sacc += S[j] * sa[j];

    const float vi = sv[i];
    float yacc = 0.0f;
#pragma unroll
    for (int j = 0; j < 64; ++j) {
      float s = S[j] * sd[j] + sacc * sb[j] + vi * sk[j];
      S[j] = s;
      yacc += s * sr[j];
    }
    y[o + i] = yacc;
    __syncthreads();
  }
}

// ------------------------------------------------------------------
// Per-head GroupNorm + (r.k.r_k) residual, emit bf16 y for output GEMM.
// One 64-thread block per (b,t,h).
// ------------------------------------------------------------------
__global__ void k_headout(const float* __restrict__ y, const float* __restrict__ r,
                          const float* __restrict__ kfin, const float* __restrict__ vmix,
                          const float* __restrict__ r_k, const float* __restrict__ ln_g,
                          const float* __restrict__ ln_b, __bf16* __restrict__ ybf) {
  __shared__ float red[64];
  const int bt = blockIdx.x / H_N;
  const int h  = blockIdx.x % H_N;
  const int hs = threadIdx.x;
  const int c  = h * HS_N + hs;
  const size_t idx = (size_t)bt * C_SZ + c;

  float yv = y[idx];
  float mu  = head_reduce64(yv, red, hs) * (1.0f / HS_N);
  float ex2 = head_reduce64(yv * yv, red, hs) * (1.0f / HS_N);
  float var = ex2 - mu * mu;
  float yn = (yv - mu) * rsqrtf(var + EPS_F) * ln_g[c] + ln_b[c];

  float rv = r[idx], kv = kfin[idx];
  float rk = head_reduce64(rv * kv * r_k[h * HS_N + hs], red, hs);

  ybf[idx] = (__bf16)(yn + rk * vmix[idx]);
}

// ------------------------------------------------------------------
// host side
// ------------------------------------------------------------------
extern "C" void kernel_launch(void* const* d_in, const int* in_sizes, int n_in,
                              void* d_out, int out_size, void* d_ws, size_t ws_size,
                              hipStream_t stream) {
  const float* x       = (const float*)d_in[0];
  const float* v_first = (const float*)d_in[1];
  const float* amask   = (const float*)d_in[2];
  const float* Wr = (const float*)d_in[3];
  const float* Wk = (const float*)d_in[4];
  const float* Wv = (const float*)d_in[5];
  const float* Wo = (const float*)d_in[6];
  const float* w0 = (const float*)d_in[7];
  const float* w1 = (const float*)d_in[8];
  const float* w2 = (const float*)d_in[9];
  const float* a0 = (const float*)d_in[10];
  const float* a1 = (const float*)d_in[11];
  const float* a2 = (const float*)d_in[12];
  const float* v0 = (const float*)d_in[13];
  const float* v1 = (const float*)d_in[14];
  const float* v2 = (const float*)d_in[15];
  const float* k_k = (const float*)d_in[16];
  const float* k_a = (const float*)d_in[17];
  const float* r_k = (const float*)d_in[18];
  const float* ln_g = (const float*)d_in[19];
  const float* ln_b = (const float*)d_in[20];
  float* out = (float*)d_out;

  // ---- carve workspace ----
  char* p = (char*)d_ws;
  auto carve = [&](size_t bytes) -> void* {
    void* q = (void*)p;
    p += (bytes + 255) & ~(size_t)255;
    return q;
  };
  const size_t NE = (size_t)M_SZ * C_SZ;          // 4M elements
  __bf16* xbf   = (__bf16*)carve(NE * 2);
  __bf16* Wr_b  = (__bf16*)carve((size_t)C_SZ * C_SZ * 2);
  __bf16* Wk_b  = (__bf16*)carve((size_t)KV_C * C_SZ * 2);
  __bf16* Wv_b  = (__bf16*)carve((size_t)KV_C * C_SZ * 2);
  __bf16* Wo_b  = (__bf16*)carve((size_t)C_SZ * C_SZ * 2);
  __bf16* w1t   = (__bf16*)carve((size_t)DD_N * C_SZ * 2);
  __bf16* a1t   = (__bf16*)carve((size_t)DD_N * C_SZ * 2);
  __bf16* v1t   = (__bf16*)carve((size_t)DMV_N * C_SZ * 2);
  __bf16* w2t   = (__bf16*)carve((size_t)C_SZ * DD_N * 2);
  __bf16* a2t   = (__bf16*)carve((size_t)C_SZ * DD_N * 2);
  __bf16* v2t   = (__bf16*)carve((size_t)C_SZ * DMV_N * 2);
  float* r_f    = (float*)carve(NE * 4);
  float* ksm    = (float*)carve((size_t)M_SZ * KV_C * 4);
  float* vsm    = (float*)carve((size_t)M_SZ * KV_C * 4);
  float* hw     = (float*)carve((size_t)M_SZ * DD_N * 4);
  float* ha     = (float*)carve((size_t)M_SZ * DD_N * 4);
  float* hv     = (float*)carve((size_t)M_SZ * DMV_N * 4);
  __bf16* hwbf  = (__bf16*)carve((size_t)M_SZ * DD_N * 2);
  __bf16* habf  = (__bf16*)carve((size_t)M_SZ * DD_N * 2);
  __bf16* hvbf  = (__bf16*)carve((size_t)M_SZ * DMV_N * 2);
  float* wpre   = (float*)carve(NE * 4);   // -> decay (in place)
  float* apre   = (float*)carve(NE * 4);   // -> aa = -kk (in place)
  float* vpre   = (float*)carve(NE * 4);   // -> v mixed (in place)
  float* kfin   = (float*)carve(NE * 4);
  float* bbuf   = (float*)carve(NE * 4);
  float* ybuf   = (float*)carve(NE * 4);
  __bf16* ybf   = (__bf16*)carve(NE * 2);
  (void)ws_size; (void)n_in; (void)in_sizes; (void)out_size;

  auto cvt = [&](const float* in, __bf16* o, int n, int do_tanh) {
    k_cvt_bf16<<<(n + 255) / 256, 256, 0, stream>>>(in, o, n, do_tanh);
  };
  auto trn = [&](const float* in, __bf16* o, int R, int Cc) {
    k_trans_bf16<<<(R * Cc + 255) / 256, 256, 0, stream>>>(in, o, R, Cc);
  };
  // branch-free dispatch: wave tile 32x64 (NSUB=4) when 64 | N, else 32x32 (NSUB=2)
  auto gemm = [&](const __bf16* A, const __bf16* W, float* Co, int M, int N, int K) {
    if (N % 64 == 0) {
      dim3 g(M / 128, N / 64);
      k_gemm_bf16<2, 4><<<g, 128, 0, stream>>>(A, W, Co, M, N, K);
    } else {            // N == 32
      dim3 g(M / 128, N / 32);
      k_gemm_bf16<2, 2><<<g, 128, 0, stream>>>(A, W, Co, M, N, K);
    }
  };

  // ---- stage 0: convert inputs/weights to bf16 ----
  cvt(x, xbf, (int)NE, 0);
  cvt(Wr, Wr_b, C_SZ * C_SZ, 0);
  cvt(Wk, Wk_b, KV_C * C_SZ, 0);
  cvt(Wv, Wv_b, KV_C * C_SZ, 0);
  cvt(Wo, Wo_b, C_SZ * C_SZ, 0);
  trn(w1, w1t, C_SZ, DD_N);    // (1024,64)  -> (64,1024)
  trn(a1, a1t, C_SZ, DD_N);
  trn(v1, v1t, C_SZ, DMV_N);   // (1024,32)  -> (32,1024)
  trn(w2, w2t, DD_N, C_SZ);    // (64,1024)  -> (1024,64)
  trn(a2, a2t, DD_N, C_SZ);
  trn(v2, v2t, DMV_N, C_SZ);   // (32,1024)  -> (1024,32)

  // ---- stage 1: projections (WMMA) ----
  gemm(xbf, Wr_b, r_f, M_SZ, C_SZ, C_SZ);      // r = x @ Wr.T
  gemm(xbf, Wk_b, ksm, M_SZ, KV_C, C_SZ);      // k (kv heads)
  gemm(xbf, Wv_b, vsm, M_SZ, KV_C, C_SZ);      // v (kv heads)
  gemm(xbf, w1t, hw, M_SZ, DD_N, C_SZ);        // x @ w1
  gemm(xbf, a1t, ha, M_SZ, DD_N, C_SZ);        // x @ a1
  gemm(xbf, v1t, hv, M_SZ, DMV_N, C_SZ);       // x @ v1

  // ---- stage 2: activations + second low-rank GEMMs (WMMA) ----
  cvt(hw, hwbf, M_SZ * DD_N, 1);               // tanh then bf16
  cvt(ha, habf, M_SZ * DD_N, 0);
  cvt(hv, hvbf, M_SZ * DMV_N, 0);
  gemm(hwbf, w2t, wpre, M_SZ, C_SZ, DD_N);     // tanh(x@w1) @ w2
  gemm(habf, a2t, apre, M_SZ, C_SZ, DD_N);     // (x@a1) @ a2
  gemm(hvbf, v2t, vpre, M_SZ, C_SZ, DMV_N);    // (x@v1) @ v2

  // ---- stage 3: gates / GQA repeat / kk-normalize ----
  k_mix<<<M_SZ * H_N, HS_N, 0, stream>>>(ksm, vsm, v_first, wpre, apre, vpre,
                                         kfin, bbuf, w0, a0, v0, k_k, k_a);

  // ---- stage 4: sequential delta-rule recurrence ----
  k_wkv7<<<B_SZ * H_N, HS_N, 0, stream>>>(r_f, wpre, kfin, vpre, apre, bbuf,
                                          amask, ybuf);

  // ---- stage 5: GroupNorm + rk residual, emit bf16 ----
  k_headout<<<M_SZ * H_N, HS_N, 0, stream>>>(ybuf, r_f, kfin, vpre, r_k,
                                             ln_g, ln_b, ybf);

  // ---- stage 6: output projection (WMMA) ----
  gemm(ybf, Wo_b, out, M_SZ, C_SZ, C_SZ);      // out = y @ Wo.T

  // ---- second tuple element: v_first passthrough ----
  hipMemcpyAsync(out + NE, v_first, NE * sizeof(float),
                 hipMemcpyDeviceToDevice, stream);
}
